// WorkingMemoryLSTMCell_80350248173768
// MI455X (gfx1250) — matrix operand order repaired and checked by reading, not compiled, not run on hardware
//
#include <hip/hip_runtime.h>
#include <hip/hip_bf16.h>
#include <cstdint>
#include <cstddef>

// Problem sizes (match reference): B=8192, IN=H=K=2048
#define BDIM 8192
#define HDIM 2048
#define KDIM 2048

// LDS staging geometry: per k-step tile = 11 weights x 16 rows x 64B,
// row stride padded to 80B for conflict-free ds_load_b128.
#define ROWB      80
#define BUFBYTES  (11 * 16 * ROWB)   // 14080
#define NCHUNK    (11 * 16 * 4)      // 704 16-byte chunks per stage

typedef __attribute__((ext_vector_type(16))) __bf16 bf16x16;
typedef __attribute__((ext_vector_type(8)))  float  f32x8;

union FragAB {
    bf16x16 v;
    uint4   q[2];   // 32 bytes
};

__device__ __forceinline__ uint16_t f32_to_bf16_rne(float f) {
    uint32_t u = __float_as_uint(f);
    uint32_t lsb = (u >> 16) & 1u;
    u += 0x7fffu + lsb;
    return (uint16_t)(u >> 16);
}

// Vectorized f32 -> bf16 conversion, 4 elements per thread.
__global__ void cvt_f32_to_bf16_kernel(const float* __restrict__ src,
                                       uint16_t* __restrict__ dst, int n4) {
    int i = blockIdx.x * blockDim.x + threadIdx.x;
    if (i < n4) {
        float4 f = ((const float4*)src)[i];
        ushort4 o;
        o.x = f32_to_bf16_rne(f.x);
        o.y = f32_to_bf16_rne(f.y);
        o.z = f32_to_bf16_rne(f.z);
        o.w = f32_to_bf16_rne(f.w);
        ((ushort4*)dst)[i] = o;
    }
}

struct Params {
    const uint16_t* Xc;       // [B][K] bf16
    const uint16_t* Hc;       // [B][K] bf16
    const uint16_t* Cc;       // [B][K] bf16
    const uint16_t* W[11];    // each [H][K] bf16: Wxi,Whi,Wci,Wxf,Whf,Wcf,Wxo,Who,Wco,Wxg,Whg
    const float*    bias[11]; // each [H] f32, same order
    const float*    c_in;     // [B][H] f32
    float*          h_out;    // [B][H] f32
    float*          c_out;    // [B][H] f32
};

__device__ __forceinline__ f32x8 wmma_bf16(const FragAB& a, const FragAB& b, f32x8 acc) {
    return __builtin_amdgcn_wmma_f32_16x16x32_bf16(
        /*neg_a=*/false, a.v, /*neg_b=*/false, b.v,
        /*c_mod=*/(short)0, acc, /*reuse_a=*/false, /*reuse_b=*/false);
}

__device__ __forceinline__ float sigmoidf_fast(float x) {
    return 1.0f / (1.0f + __expf(-x));
}

// Block = 256 threads = 8 waves stacked in M (128 rows) sharing one n-tile (16 cols).
// The 11 weight B-tiles for each 32-wide k-step are staged into double-buffered LDS
// with GLOBAL_LOAD_ASYNC_TO_LDS_B128 (ASYNCcnt), overlapping the WMMA of step s with
// the staging of step s+1. B-fragments are pulled from LDS group-ahead (c-path,
// x-path, h-path) so every wmma group runs behind a partial DScnt wait.
// amdgpu_waves_per_eu(2): 8-wave blocks already imply 2 waves/SIMD residency, so
// raise the per-wave VGPR budget to match and let the scheduler keep fragments live.
__global__ void __launch_bounds__(256)
__attribute__((amdgpu_waves_per_eu(2)))
lstm_wmma_fused_kernel(Params p) {
    __shared__ __align__(16) uint8_t smem[2][BUFBYTES];

    const int lane   = threadIdx.x & 31;
    const int warp   = threadIdx.x >> 5;
    const int n_base = blockIdx.x * 16;            // HDIM/16 = 128 n-tiles
    const int m_base = blockIdx.y * 128 + warp * 16;
    const int row    = lane & 15;
    const int half   = lane >> 4;

    // A-fragment per-lane base: row M=row; K chunks {half*8..+7} and {16+half*8..+7}
    const uint16_t* ax = p.Xc + (size_t)(m_base + row) * KDIM + half * 8;
    const uint16_t* ah = p.Hc + (size_t)(m_base + row) * KDIM + half * 8;
    const uint16_t* ac = p.Cc + (size_t)(m_base + row) * KDIM + half * 8;

    const uint32_t lds_base = (uint32_t)(uintptr_t)(&smem[0][0]);

    // Stage the 11 B-tiles for k-step s into LDS buffer `buf` (async, ASYNCcnt).
    // Every thread issues exactly 3 async b128 ops (tail clamped -> duplicate
    // benign copy) so ASYNCcnt bookkeeping is uniform across all 8 waves.
    auto stage = [&](int s, int buf) {
        const int k = s * 32;
#pragma unroll
        for (int it = 0; it < 3; ++it) {
            int t = (int)threadIdx.x + it * 256;
            if (t >= NCHUNK) t = NCHUNK - 1;
            const int j  = t >> 6;          // weight index 0..10
            const int rw = (t >> 2) & 15;   // row (N col) 0..15
            const int cp = t & 3;           // 16B chunk within 64B row
            const uint16_t* g = p.W[j] + (size_t)(n_base + rw) * KDIM + k + cp * 8;
            const uint32_t  l = lds_base + (uint32_t)(buf * BUFBYTES
                                + (j * 16 + rw) * ROWB + cp * 16);
            asm volatile("global_load_async_to_lds_b128 %0, %1, off"
                         :: "v"(l), "v"(g) : "memory");
        }
    };

    f32x8 acc_i  = {}, acc_f  = {}, acc_o  = {}, acc_g = {};
    f32x8 acc_ci = {}, acc_cf = {}, acc_co = {};

    stage(0, 0);

    const int NSTEP = KDIM / 32;  // 64
    for (int s = 0; s < NSTEP; ++s) {
        const int cur = s & 1;
        const int k = s * 32;

        // Issue A-fragment global loads early: their latency hides behind the
        // async wait + barrier below (LOADcnt is independent of ASYNCcnt/DScnt).
        FragAB a_x, a_h, a_c;
        a_x.q[0] = *(const uint4*)(ax + k);
        a_x.q[1] = *(const uint4*)(ax + k + 16);
        a_h.q[0] = *(const uint4*)(ah + k);
        a_h.q[1] = *(const uint4*)(ah + k + 16);
        a_c.q[0] = *(const uint4*)(ac + k);
        a_c.q[1] = *(const uint4*)(ac + k + 16);

        if (s + 1 < NSTEP) {
            stage(s + 1, cur ^ 1);
            // my stage(s) writes complete; stage(s+1)'s 3 ops still in flight
            asm volatile("s_wait_asynccnt 0x3" ::: "memory");
        } else {
            asm volatile("s_wait_asynccnt 0x0" ::: "memory");
        }
        __syncthreads();  // all waves' stage(s) data visible in LDS

        const uint8_t* bufp = &smem[cur][0];
#define LOADG(dst, j) do { \
        const uint4* pb = (const uint4*)(bufp + ((j) * 16 + row) * ROWB + half * 16); \
        (dst).q[0] = pb[0]; (dst).q[1] = pb[2]; } while (0)

        // Group-ahead pipeline: loads for group g+1 are in flight while the
        // wmma burst of group g executes -> partial DScnt waits, no full drains.
        FragAB b_ci, b_cf, b_co;
        LOADG(b_ci, 2); LOADG(b_cf, 5); LOADG(b_co, 8);

        FragAB b_xi, b_xf, b_xo, b_xg;
        LOADG(b_xi, 0); LOADG(b_xf, 3); LOADG(b_xo, 6); LOADG(b_xg, 9);

        // c-path: tanh(c @ Wc^T) pre-activations (x-path loads still streaming)
        acc_ci = wmma_bf16(a_c, b_ci, acc_ci);
        acc_cf = wmma_bf16(a_c, b_cf, acc_cf);
        acc_co = wmma_bf16(a_c, b_co, acc_co);

        FragAB b_hi, b_hf, b_ho, b_hg;
        LOADG(b_hi, 1); LOADG(b_hf, 4); LOADG(b_ho, 7); LOADG(b_hg, 10);

        // x-path (h-path loads still streaming)
        acc_i  = wmma_bf16(a_x, b_xi, acc_i);
        acc_f  = wmma_bf16(a_x, b_xf, acc_f);
        acc_o  = wmma_bf16(a_x, b_xo, acc_o);
        acc_g  = wmma_bf16(a_x, b_xg, acc_g);

        // h-path (its loads had the whole x-path wmma burst to complete)
        acc_i  = wmma_bf16(a_h, b_hi, acc_i);
        acc_f  = wmma_bf16(a_h, b_hf, acc_f);
        acc_o  = wmma_bf16(a_h, b_ho, acc_o);
        acc_g  = wmma_bf16(a_h, b_hg, acc_g);
#undef LOADG

        __syncthreads();  // everyone done reading buf `cur` before it is re-staged
    }

    // Epilogue: C/D layout -> VGPR r holds (M = r + 8*half, N = lane&15)
    const int gn = n_base + row;
    const float bxi = p.bias[0][gn], bhi = p.bias[1][gn], bci = p.bias[2][gn];
    const float bxf = p.bias[3][gn], bhf = p.bias[4][gn], bcf = p.bias[5][gn];
    const float bxo = p.bias[6][gn], bho = p.bias[7][gn], bco = p.bias[8][gn];
    const float bxg = p.bias[9][gn], bhg = p.bias[10][gn];

#pragma unroll
    for (int r = 0; r < 8; ++r) {
        const int gm = m_base + r + half * 8;
        const size_t idx = (size_t)gm * HDIM + gn;
        const float c_old = p.c_in[idx];
        const float tci = tanhf(acc_ci[r] + bci);
        const float tcf = tanhf(acc_cf[r] + bcf);
        const float tco = tanhf(acc_co[r] + bco);
        const float it = sigmoidf_fast(acc_i[r] + bxi + bhi + tci);
        const float ft = sigmoidf_fast(acc_f[r] + bxf + bhf + tcf);
        const float ot = sigmoidf_fast(acc_o[r] + bxo + bho + tco);
        const float gt = tanhf(acc_g[r] + bxg + bhg);
        const float cn = ft * c_old + it * gt;
        const float hn = ot * tanhf(cn);
        p.h_out[idx] = hn;
        p.c_out[idx] = cn;
    }
}

extern "C" void kernel_launch(void* const* d_in, const int* in_sizes, int n_in,
                              void* d_out, int out_size, void* d_ws, size_t ws_size,
                              hipStream_t stream) {
    (void)in_sizes; (void)n_in; (void)out_size; (void)ws_size;

    const float* x = (const float*)d_in[0];
    const float* h = (const float*)d_in[1];
    const float* c = (const float*)d_in[2];
    // d_in[3 + 2*j] = weight j, d_in[4 + 2*j] = bias j, in order:
    // Wxi,Whi,Wci,Wxf,Whf,Wcf,Wxo,Who,Wco,Wxg,Whg

    uint8_t* ws = (uint8_t*)d_ws;
    size_t off = 0;
    uint16_t* Xc = (uint16_t*)(ws + off); off += (size_t)BDIM * KDIM * 2;
    uint16_t* Hc = (uint16_t*)(ws + off); off += (size_t)BDIM * KDIM * 2;
    uint16_t* Cc = (uint16_t*)(ws + off); off += (size_t)BDIM * KDIM * 2;
    uint16_t* Wc[11];
    for (int j = 0; j < 11; ++j) {
        Wc[j] = (uint16_t*)(ws + off);
        off += (size_t)HDIM * KDIM * 2;
    }
    // Total workspace used: 3*32MB + 11*8MB = 184 MB

    auto cvt = [&](const float* s, uint16_t* d, size_t n) {
        int n4 = (int)(n / 4);
        int blocks = (n4 + 255) / 256;
        cvt_f32_to_bf16_kernel<<<blocks, 256, 0, stream>>>(s, d, n4);
    };
    cvt(x, Xc, (size_t)BDIM * KDIM);
    cvt(h, Hc, (size_t)BDIM * KDIM);
    cvt(c, Cc, (size_t)BDIM * KDIM);
    for (int j = 0; j < 11; ++j)
        cvt((const float*)d_in[3 + 2 * j], Wc[j], (size_t)HDIM * KDIM);

    Params p;
    p.Xc = Xc; p.Hc = Hc; p.Cc = Cc;
    for (int j = 0; j < 11; ++j) {
        p.W[j]    = Wc[j];
        p.bias[j] = (const float*)d_in[4 + 2 * j];
    }
    p.c_in  = c;
    p.h_out = (float*)d_out;
    p.c_out = (float*)d_out + (size_t)BDIM * HDIM;

    // Grid: 128 n-tiles x 64 m-strips of 128 rows (8 waves each)
    dim3 grid(HDIM / 16, BDIM / 128);
    lstm_wmma_fused_kernel<<<grid, 256, 0, stream>>>(p);
}
